// TransActor_12489764897111
// MI455X (gfx1250) — matrix operand order repaired and checked by reading, not compiled, not run on hardware
//
#include <hip/hip_runtime.h>
#include <hip/hip_bf16.h>

typedef _Float16 half_t;
typedef __attribute__((ext_vector_type(16))) _Float16 v16h;
typedef __attribute__((ext_vector_type(8)))  float    v8f;

#define NN    50000
#define EE    100000
#define FIN   20
#define HID   500
#define HEADS 4
#define DD    2000
#define OUTC  2
#define EPSF  1e-5f
#define NB    8000   // q|k|v|s concatenated output columns

#ifndef __has_builtin
#define __has_builtin(x) 0
#endif
#if __has_builtin(__builtin_amdgcn_tensor_load_to_lds) && __has_builtin(__builtin_amdgcn_s_wait_tensorcnt)
#define HAVE_TDM 1
#else
#define HAVE_TDM 0
#endif
#if __has_builtin(__builtin_amdgcn_global_load_async_to_lds_b128) && __has_builtin(__builtin_amdgcn_s_wait_asynccnt)
#define HAVE_ASYNC 1
#else
#define HAVE_ASYNC 0
#endif

#if HAVE_ASYNC
// Builtin takes pointers to 16-byte int vectors: src in global (AS1), dst in LDS (AS3).
typedef __attribute__((__vector_size__(16))) int i128v;
typedef __attribute__((address_space(1))) i128v as1_i128;
typedef __attribute__((address_space(3))) i128v as3_i128;
#endif

// ---------------- helpers ----------------
__device__ __forceinline__ unsigned fenc(float f) {
  unsigned u = __float_as_uint(f);
  return (u & 0x80000000u) ? ~u : (u | 0x80000000u);
}
__device__ __forceinline__ float fdec(unsigned u) {
  u = (u & 0x80000000u) ? (u & 0x7FFFFFFFu) : ~u;
  return __uint_as_float(u);
}

// ---------------- cast / pack kernels ----------------
__global__ void cast_pad_f16(const float* __restrict__ in, half_t* __restrict__ out,
                             int M, int K, int Kp) {
  long long i = (long long)blockIdx.x * blockDim.x + threadIdx.x;
  if (i >= (long long)M * Kp) return;
  int r = (int)(i / Kp), c = (int)(i % Kp);
  out[i] = (c < K) ? (half_t)in[(size_t)r * K + c] : (half_t)0.0f;
}

// W16T is COLUMN-major: W16T[j][k] (j = 0..7999 over q|k|v|s, k contiguous 0..Kp-1)
__global__ void pack_weights(const float* __restrict__ Wq, const float* __restrict__ Wk,
                             const float* __restrict__ Wv, const float* __restrict__ Ws,
                             half_t* __restrict__ out, int K, int Kp) {
  long long i = (long long)blockIdx.x * blockDim.x + threadIdx.x;   // i = j*Kp + k
  if (i >= (long long)Kp * NB) return;
  int j = (int)(i / Kp), k = (int)(i % Kp);
  float v = 0.0f;
  if (k < K) {
    int sel = j / DD, d = j % DD;
    const float* W = (sel == 0) ? Wq : (sel == 1) ? Wk : (sel == 2) ? Wv : Ws;
    v = W[(size_t)k * DD + d];
  }
  out[i] = (half_t)v;
}

__global__ void pack_bias(const float* __restrict__ bq, const float* __restrict__ bk,
                          const float* __restrict__ bv, const float* __restrict__ bs,
                          float* __restrict__ out) {
  int j = blockIdx.x * blockDim.x + threadIdx.x;
  if (j >= NB) return;
  int sel = j / DD, d = j % DD;
  const float* b = (sel == 0) ? bq : (sel == 1) ? bk : (sel == 2) ? bv : bs;
  out[j] = b[d];
}

// ---------------- WMMA GEMM ----------------
// C[M,8000] = A[M,Kp](f16) @ W(f16) + bias, W given column-major [8000][Kp]
// cols 0..5999 (q|k|v) -> Cqkv[M,6000]; cols 6000..7999 (s) -> Xout[M,2000]
#define BM 256
#define BN 64
#define BK 64
// 8 waves; wave w computes rows [w*32, w*32+32) x all 64 cols; K staged 64 per iter
// => 2 (M) x 4 (N) x 2 (K substeps) = 16 v_wmma per loop body

#if HAVE_TDM
typedef unsigned int v4u __attribute__((ext_vector_type(4)));
typedef int v8i __attribute__((ext_vector_type(8)));
typedef int v4i __attribute__((ext_vector_type(4)));

// TDM load of a BM x BK f16 tile from a row-major [Mrem, Kp] region into LDS.
// D# per CDNA5 ISA 8.3/8.4: group0 = {count=1, lds_addr, global_addr, type=2},
// group1 = {data_size=2B, tensor_dim0=Kp, tensor_dim1=Mrem, tile BKxBM, stride=Kp}.
__device__ __forceinline__ void tdm_load_tile_a(const half_t* gtile, unsigned lds_off,
                                                int Kp, int Mrem) {
  unsigned long long ga = (unsigned long long)(uintptr_t)gtile;
  v4u g0;
  g0.x = 1u;                                            // count=1, user descriptor
  g0.y = lds_off;                                       // lds_addr (bytes)
  g0.z = (unsigned)(ga & 0xFFFFFFFFull);                // global_addr[31:0]
  g0.w = (unsigned)((ga >> 32) & 0x01FFFFFFull) | (2u << 30); // addr[56:32] | type=2
  v8i g1;
  g1[0] = (1 << 16);                                    // data_size = 1 (2 bytes)
  g1[1] = (int)((unsigned)(Kp & 0xFFFF) << 16);         // tensor_dim0[15:0] @ bits 63:48
  g1[2] = (int)(((unsigned)Kp >> 16) |                  // tensor_dim0[31:16]
                ((unsigned)(Mrem & 0xFFFF) << 16));     // tensor_dim1[15:0]
  g1[3] = (int)(((unsigned)Mrem >> 16) |                // tensor_dim1[31:16]
                ((unsigned)BK << 16));                  // tile_dim0 = BK
  g1[4] = BM;                                           // tile_dim1 = BM, tile_dim2 = 0
  g1[5] = Kp;                                           // tensor_dim0_stride[31:0]
  g1[6] = 0;                                            // stride hi | dim1_stride lo
  g1[7] = 0;
  v4i z4 = {0, 0, 0, 0};
#if __clang_major__ >= 23
  v8i z8 = {0, 0, 0, 0, 0, 0, 0, 0};
  __builtin_amdgcn_tensor_load_to_lds(g0, g1, z4, z4, z8, 0);
#else
  __builtin_amdgcn_tensor_load_to_lds(g0, g1, z4, z4, 0);
#endif
}
#endif

__global__ __launch_bounds__(256)
void wmma_gemm_qkvs(const half_t* __restrict__ A, int lda,
                    const half_t* __restrict__ BT,   // column-major [NB][Kp]
                    const float* __restrict__ bias,
                    float* __restrict__ Cqkv,
                    float* __restrict__ Xout,
                    int M, int Kp) {
  __shared__ __align__(16) half_t sA[BM][BK];   // 32 KB (TDM destination)
  __shared__ __align__(16) half_t sBT[BN][BK];  // 8 KB, [n][k]

  const int tid  = threadIdx.x;
  const int wave = tid >> 5;
  const int lane = tid & 31;
  const int blockM = blockIdx.y * BM;
  const int blockN = blockIdx.x * BN;
  const int l  = lane & 15;
  const int hi = lane >> 4;

  v8f acc[2][4] = {};

#if HAVE_TDM
  const unsigned sA_off = (unsigned)(uintptr_t)(void*)&sA[0][0];
  int mrem = M - blockM; if (mrem < 0) mrem = 0;
#endif

  for (int kb = 0; kb < Kp; kb += BK) {
#if HAVE_TDM
    // --- A tile via Tensor Data Mover: one wave issues, OOB rows return zero ---
    if (tid == 0)
      tdm_load_tile_a(A + (size_t)blockM * lda + kb, sA_off, Kp, mrem);
#else
    // --- A tile: 256x64 halfs, 2048 chunks of 8 halfs, 8 per thread ---
    for (int c = 0; c < 8; ++c) {
      int chunk = tid + c * 256;
      int row   = chunk >> 3;
      int col8  = (chunk & 7) * 8;
      int grow  = blockM + row;
      if (grow < M) {
        *(float4*)(&sA[row][col8]) =
            *(const float4*)(A + (size_t)grow * lda + kb + col8);
      } else {
        float4 z; z.x = z.y = z.z = z.w = 0.0f;
        *(float4*)(&sA[row][col8]) = z;
      }
    }
#endif
    // --- B tile: straight block copy (column-major matches LDS layout) ---
    // 64 cols x 64 halfs = 512 chunks of 8 halfs, 2 per thread
    for (int c = 0; c < 2; ++c) {
      int chunk = tid + c * 256;
      int n  = chunk >> 3;
      int k8 = (chunk & 7) * 8;
      const half_t* gp = BT + (size_t)(blockN + n) * Kp + kb + k8;
      half_t* lp = &sBT[n][k8];
#if HAVE_ASYNC
      __builtin_amdgcn_global_load_async_to_lds_b128(
          (as1_i128*)(uintptr_t)gp, (as3_i128*)(uintptr_t)lp, 0, 0);
#else
      *(float4*)lp = *(const float4*)gp;
#endif
    }
#if HAVE_ASYNC
    __builtin_amdgcn_s_wait_asynccnt(0);
#endif
#if HAVE_TDM
    if (tid == 0) __builtin_amdgcn_s_wait_tensorcnt(0);
#endif
    __syncthreads();

    // --- fragments + WMMA: two K=32 substeps; B frags reused across A frags ---
#pragma unroll
    for (int kk = 0; kk < BK; kk += 32) {
      v16h bf[4];
#pragma unroll
      for (int j = 0; j < 4; ++j) {
        const half_t* bp = &sBT[j * 16 + l][kk + hi * 16];
        *(float4*)(&bf[j])         = *(const float4*)(bp);
        *(((float4*)(&bf[j])) + 1) = *(const float4*)(bp + 8);
      }
#pragma unroll
      for (int i = 0; i < 2; ++i) {
        int rowm = wave * 32 + i * 16 + l;
        v16h afrag;
        const half_t* ap = &sA[rowm][kk];
        *(float4*)&afrag         = *(const float4*)(ap + hi * 16);
        *(((float4*)&afrag) + 1) = *(const float4*)(ap + hi * 16 + 8);
#pragma unroll
        for (int j = 0; j < 4; ++j)
          acc[i][j] = __builtin_amdgcn_wmma_f32_16x16x32_f16(
              false, afrag, false, bf[j], (short)0, acc[i][j], false, false);
      }
    }
    __syncthreads();
  }

  // --- epilogue: D layout N=lane%16, M = 8*hi + vgpr ---
  for (int i = 0; i < 2; ++i)
    for (int j = 0; j < 4; ++j) {
      int colg = blockN + j * 16 + l;
      float bb = bias[colg];
      for (int g = 0; g < 8; ++g) {
        int rowg = blockM + wave * 32 + i * 16 + hi * 8 + g;
        if (rowg < M) {
          float val = acc[i][j][g] + bb;
          if (colg < 3 * DD) Cqkv[(size_t)rowg * (3 * DD) + colg] = val;
          else               Xout[(size_t)rowg * DD + (colg - 3 * DD)] = val;
        }
      }
    }
}

// ---------------- edge / attention kernels ----------------
__global__ void init_seg(unsigned* __restrict__ m, float* __restrict__ denom) {
  int i = blockIdx.x * blockDim.x + threadIdx.x;
  if (i >= NN * HEADS) return;
  m[i] = fenc(-INFINITY);
  denom[i] = 0.0f;
}

__global__ void edge_alpha_max(const int* __restrict__ ei, const float* __restrict__ qkv,
                               float* __restrict__ alpha, unsigned* __restrict__ m) {
  int gw   = (blockIdx.x * blockDim.x + threadIdx.x) >> 5;
  int lane = threadIdx.x & 31;
  if (gw >= EE * HEADS) return;
  int e = gw / HEADS, h = gw % HEADS;
  int src = ei[e], dst = ei[EE + e];
  const float* q = qkv + (size_t)dst * (3 * DD) + h * HID;          // q block
  const float* k = qkv + (size_t)src * (3 * DD) + DD + h * HID;     // k block
  float s = 0.0f;
  for (int c = lane; c < HID; c += 32) s += q[c] * k[c];
  for (int off = 16; off > 0; off >>= 1) s += __shfl_down(s, off, 32);
  if (lane == 0) {
    s *= rsqrtf((float)HID);
    alpha[(size_t)e * HEADS + h] = s;
    atomicMax(&m[dst * HEADS + h], fenc(s));
  }
}

__global__ void edge_exp_sum(const int* __restrict__ ei, const float* __restrict__ alpha,
                             const unsigned* __restrict__ m, float* __restrict__ ex,
                             float* __restrict__ denom) {
  int i = blockIdx.x * blockDim.x + threadIdx.x;
  if (i >= EE * HEADS) return;
  int e = i / HEADS, h = i % HEADS;
  int dst = ei[EE + e];
  float v = expf(alpha[i] - fdec(m[dst * HEADS + h]));
  ex[i] = v;
  unsafeAtomicAdd(&denom[dst * HEADS + h], v);
}

__global__ void edge_aggregate(const int* __restrict__ ei, const float* __restrict__ qkv,
                               const float* __restrict__ ex, const float* __restrict__ denom,
                               float* __restrict__ xout) {
  int gw   = (blockIdx.x * blockDim.x + threadIdx.x) >> 5;
  int lane = threadIdx.x & 31;
  if (gw >= EE * HEADS) return;
  int e = gw / HEADS, h = gw % HEADS;
  int src = ei[e], dst = ei[EE + e];
  float w = ex[(size_t)e * HEADS + h] / denom[dst * HEADS + h];
  const float* v = qkv + (size_t)src * (3 * DD) + 2 * DD + h * HID;  // v block
  float* o = xout + (size_t)dst * DD + h * HID;
  for (int c = lane; c < HID; c += 32) unsafeAtomicAdd(&o[c], w * v[c]);
}

// ---------------- batch norm + head ----------------
__global__ void zero_bn(float* __restrict__ s, float* __restrict__ s2) {
  int i = blockIdx.x * blockDim.x + threadIdx.x;
  if (i < DD) { s[i] = 0.0f; s2[i] = 0.0f; }
}

#define BN_CHUNK 200
__global__ void bn_stats(const float* __restrict__ x, float* __restrict__ sum,
                         float* __restrict__ sumsq) {
  int col = blockIdx.x * blockDim.x + threadIdx.x;
  if (col >= DD) return;
  int r0 = blockIdx.y * BN_CHUNK;
  int r1 = min(r0 + BN_CHUNK, NN);
  float s = 0.0f, s2 = 0.0f;
  for (int r = r0; r < r1; ++r) {
    float v = x[(size_t)r * DD + col];
    s += v; s2 += v * v;
  }
  unsafeAtomicAdd(&sum[col], s);
  unsafeAtomicAdd(&sumsq[col], s2);
}

__global__ void bn_final(const float* __restrict__ sum, const float* __restrict__ sumsq,
                         const float* __restrict__ gamma, const float* __restrict__ beta,
                         float* __restrict__ scale, float* __restrict__ shift,
                         const float* __restrict__ x, float* __restrict__ out_rsu) {
  int d = blockIdx.x * blockDim.x + threadIdx.x;
  if (d >= DD) return;
  float mean = sum[d] * (1.0f / NN);
  float var  = sumsq[d] * (1.0f / NN) - mean * mean;
  float sc   = gamma[d] * rsqrtf(var + EPSF);
  float sh   = beta[d] - mean * sc;
  scale[d] = sc; shift[d] = sh;
  out_rsu[d] = x[d] * sc + sh;   // normalized row 0 = rsu embedding
}

__global__ void head_softmax(const float* __restrict__ x, const float* __restrict__ scale,
                             const float* __restrict__ shift, const float* __restrict__ Wl,
                             const float* __restrict__ bl, float* __restrict__ out) {
  int gw   = (blockIdx.x * blockDim.x + threadIdx.x) >> 5;
  int lane = threadIdx.x & 31;
  if (gw >= NN) return;
  const float* row = x + (size_t)gw * DD;
  float a0 = 0.0f, a1 = 0.0f;
  for (int d = lane; d < DD; d += 32) {
    float xb = row[d] * scale[d] + shift[d];
    a0 += xb * Wl[d * 2 + 0];
    a1 += xb * Wl[d * 2 + 1];
  }
  for (int off = 16; off > 0; off >>= 1) {
    a0 += __shfl_down(a0, off, 32);
    a1 += __shfl_down(a1, off, 32);
  }
  if (lane == 0) {
    a0 = fmaxf(a0 + bl[0], 0.0f);
    a1 = fmaxf(a1 + bl[1], 0.0f);
    float mx = fmaxf(a0, a1);
    float e0 = __expf(a0 - mx), e1 = __expf(a1 - mx);
    float inv = 1.0f / (e0 + e1);
    out[(size_t)gw * 2 + 0] = e0 * inv;
    out[(size_t)gw * 2 + 1] = e1 * inv;
  }
}

// ---------------- host ----------------
extern "C" void kernel_launch(void* const* d_in, const int* in_sizes, int n_in,
                              void* d_out, int out_size, void* d_ws, size_t ws_size,
                              hipStream_t stream) {
  const float* x0 = (const float*)d_in[0];
  const int*   ei = (const int*)d_in[1];
  const float *Wq1 = (const float*)d_in[2],  *bq1 = (const float*)d_in[3];
  const float *Wk1 = (const float*)d_in[4],  *bk1 = (const float*)d_in[5];
  const float *Wv1 = (const float*)d_in[6],  *bv1 = (const float*)d_in[7];
  const float *Ws1 = (const float*)d_in[8],  *bs1 = (const float*)d_in[9];
  const float *Wq2 = (const float*)d_in[10], *bq2 = (const float*)d_in[11];
  const float *Wk2 = (const float*)d_in[12], *bk2 = (const float*)d_in[13];
  const float *Wv2 = (const float*)d_in[14], *bv2 = (const float*)d_in[15];
  const float *Ws2 = (const float*)d_in[16], *bs2 = (const float*)d_in[17];
  const float *gamma = (const float*)d_in[18], *beta = (const float*)d_in[19];
  const float *Wl = (const float*)d_in[20], *bl = (const float*)d_in[21];
  float* out = (float*)d_out;

  // carve workspace
  char* w = (char*)d_ws;
  auto carve = [&](size_t bytes) { void* p = (void*)w; w += (bytes + 255) & ~(size_t)255; return p; };
  half_t*   A16    = (half_t*)  carve((size_t)NN * 2048 * sizeof(half_t));
  half_t*   W16T   = (half_t*)  carve((size_t)2048 * NB * sizeof(half_t));
  float*    bcat   = (float*)   carve((size_t)NB * sizeof(float));
  float*    Cqkv   = (float*)   carve((size_t)NN * 3 * DD * sizeof(float));
  float*    Xout   = (float*)   carve((size_t)NN * DD * sizeof(float));
  float*    alpha  = (float*)   carve((size_t)EE * HEADS * sizeof(float));
  float*    exbuf  = (float*)   carve((size_t)EE * HEADS * sizeof(float));
  unsigned* menc   = (unsigned*)carve((size_t)NN * HEADS * sizeof(unsigned));
  float*    denom  = (float*)   carve((size_t)NN * HEADS * sizeof(float));
  float*    colsum = (float*)   carve((size_t)DD * sizeof(float));
  float*    colsq  = (float*)   carve((size_t)DD * sizeof(float));
  float*    scale  = (float*)   carve((size_t)DD * sizeof(float));
  float*    shift  = (float*)   carve((size_t)DD * sizeof(float));

  const int NT = 256;
  auto nb = [](long long n, int t) { return (unsigned)((n + t - 1) / t); };

  auto run_layer = [&](const float* xin, int K, int Kp,
                       const float* Wq, const float* bq, const float* Wk, const float* bk,
                       const float* Wv, const float* bv, const float* Ws, const float* bs) {
    cast_pad_f16<<<nb((long long)NN * Kp, NT), NT, 0, stream>>>(xin, A16, NN, K, Kp);
    pack_weights<<<nb((long long)Kp * NB, NT), NT, 0, stream>>>(Wq, Wk, Wv, Ws, W16T, K, Kp);
    pack_bias<<<nb(NB, NT), NT, 0, stream>>>(bq, bk, bv, bs, bcat);
    dim3 g(NB / BN, (NN + BM - 1) / BM);
    wmma_gemm_qkvs<<<g, 256, 0, stream>>>(A16, Kp, W16T, bcat, Cqkv, Xout, NN, Kp);
    init_seg<<<nb((long long)NN * HEADS, NT), NT, 0, stream>>>(menc, denom);
    edge_alpha_max<<<nb((long long)EE * HEADS * 32, NT), NT, 0, stream>>>(ei, Cqkv, alpha, menc);
    edge_exp_sum<<<nb((long long)EE * HEADS, NT), NT, 0, stream>>>(ei, alpha, menc, exbuf, denom);
    edge_aggregate<<<nb((long long)EE * HEADS * 32, NT), NT, 0, stream>>>(ei, Cqkv, exbuf, denom, Xout);
  };

  // layer 1: K=20 padded to 64 (one K iteration)
  run_layer(x0, FIN, 64, Wq1, bq1, Wk1, bk1, Wv1, bv1, Ws1, bs1);
  // layer 2: K=2000 padded to 2048 (Xout consumed into A16 before GEMM overwrites it)
  run_layer(Xout, DD, 2048, Wq2, bq2, Wk2, bk2, Wv2, bv2, Ws2, bs2);

  // batch norm over nodes
  zero_bn<<<nb(DD, NT), NT, 0, stream>>>(colsum, colsq);
  dim3 gs(nb(DD, NT), (NN + BN_CHUNK - 1) / BN_CHUNK);
  bn_stats<<<gs, NT, 0, stream>>>(Xout, colsum, colsq);
  bn_final<<<nb(DD, NT), NT, 0, stream>>>(colsum, colsq, gamma, beta, scale, shift,
                                          Xout, out + (size_t)NN * OUTC);
  // head: relu(linear) + softmax, writes action_prob [NN,2]
  head_softmax<<<nb((long long)NN * 32, NT), NT, 0, stream>>>(Xout, scale, shift, Wl, bl, out);
}
